// PerformerAttention_34505767256281
// MI455X (gfx1250) — matrix operand order repaired and checked by reading, not compiled, not run on hardware
//
#include <hip/hip_runtime.h>

// ---------------------------------------------------------------------------
// PerformerAttention for MI455X (gfx1250, wave32, WMMA 16x16x32 f16).
//   k1: qkv = x @ Wqkv          (f32 in, f16 out, ReLU fused on q/k columns)
//   k2: flash attention per head (f16 in/out, f32 accumulation + softmax)
//   k3: out = attn @ Wout + b    (f16 A, f32 out)
// Global->LDS staging is overlapped with compute via global_prefetch of the
// next tile (probe-confirmed __builtin_prefetch -> global_prefetch_b8).
// ---------------------------------------------------------------------------

typedef __attribute__((ext_vector_type(16))) _Float16 v16h;
typedef __attribute__((ext_vector_type(8)))  float    v8f;

#define NTOK  4096
#define CDIM  512
#define HEADS 16
#define DHEAD 32
#define K3    1536
#define QKSCALE 0.17677669529663687f   // 32^-0.5

// ---- WMMA fragment helpers (layouts per CDNA5 ISA 7.12.2) -----------------

// A-matrix 16x32 f16: lane L holds row (L&15); VGPR i holds K pair
// {0,2,4,6,16,18,20,22}[i] (+8 for lanes 16..31).
__device__ __forceinline__ v16h frag_a_ld(const _Float16* p, int ld) {
  const int lane = threadIdx.x & 31;
  const int row  = lane & 15;
  const int khi  = (lane >> 4) << 3;
  const _Float16* rp = p + row * ld;
  v16h a;
#pragma unroll
  for (int i = 0; i < 8; ++i) {
    const int k = ((i < 4) ? (2 * i) : (8 + 2 * i)) + khi;  // 0,2,4,6,16,18,20,22
    a[2 * i]     = rp[k];
    a[2 * i + 1] = rp[k + 1];
  }
  return a;
}

// B-matrix 32x16 f16 read from [N][K]-major storage (row stride ld halves):
// lane L holds column (L&15); element s holds K = s (s<8) / s+8 (s>=8),
// +8 for lanes 16..31 (same K order as the A fragment).
__device__ __forceinline__ v16h frag_b_ld(const _Float16* p, int ld) {
  const int lane = threadIdx.x & 31;
  const int col  = lane & 15;
  const int khi  = (lane >> 4) << 3;
  const _Float16* cp = p + col * ld + khi;
  v16h b;
#pragma unroll
  for (int i = 0; i < 16; ++i) b[i] = cp[(i < 8) ? i : (i + 8)];
  return b;
}

__device__ __forceinline__ v8f wmma_f16(v16h a, v16h b, v8f c) {
  return __builtin_amdgcn_wmma_f32_16x16x32_f16(
      /*neg_a=*/false, a, /*neg_b=*/false, b,
      /*c_mod=*/(short)0, c, /*reuse_a=*/false, /*reuse_b=*/false);
}

// ---------------------------------------------------------------------------
// Kernel 1: qkv[4096,1536] = x[4096,512] @ Wqkv[512,1536]; ReLU on cols<1024;
// result stored f16. Block tile 128x64, 8 waves, wave tile 32x32.
// ---------------------------------------------------------------------------
__global__ __launch_bounds__(256) void qkv_gemm_kernel(
    const float* __restrict__ x, const float* __restrict__ Wqkv,
    _Float16* __restrict__ qkvh) {
  __shared__ __align__(16) _Float16 As[128][40];  // [M][K] f16
  __shared__ __align__(16) _Float16 Bt[64][40];   // [N][K] f16 (transposed)

  const int tid  = threadIdx.x;
  const int lane = tid & 31;
  const int wid  = tid >> 5;
  const int m0   = blockIdx.y * 128;
  const int n0   = blockIdx.x * 64;
  const int wm   = (wid >> 1) * 32;
  const int wn   = (wid & 1) * 32;

  v8f acc[4] = {};

  for (int k0 = 0; k0 < CDIM; k0 += 32) {
    __syncthreads();
    // stage A: 128x32 f32 -> f16 (1024 float4 chunks / 256 threads)
#pragma unroll
    for (int i = 0; i < 4; ++i) {
      const int idx = tid + i * 256;
      const int r = idx >> 3, c = (idx & 7) * 4;
      const float4 v = *(const float4*)(x + (size_t)(m0 + r) * CDIM + k0 + c);
      As[r][c]     = (_Float16)v.x;
      As[r][c + 1] = (_Float16)v.y;
      As[r][c + 2] = (_Float16)v.z;
      As[r][c + 3] = (_Float16)v.w;
    }
    // stage B transposed: 32(k) x 64(n) f32 -> Bt[n][k] f16
#pragma unroll
    for (int i = 0; i < 2; ++i) {
      const int idx = tid + i * 256;
      const int kr = idx >> 4, c = (idx & 15) * 4;
      const float4 v = *(const float4*)(Wqkv + (size_t)(k0 + kr) * K3 + n0 + c);
      Bt[c][kr]     = (_Float16)v.x;
      Bt[c + 1][kr] = (_Float16)v.y;
      Bt[c + 2][kr] = (_Float16)v.z;
      Bt[c + 3][kr] = (_Float16)v.w;
    }
    // prefetch next k-tile while this one computes (global_prefetch_b8)
    if (k0 + 32 < CDIM) {
      __builtin_prefetch(x + (size_t)(m0 + (tid >> 3)) * CDIM + (k0 + 32) + (tid & 7) * 4, 0, 3);
      __builtin_prefetch(Wqkv + (size_t)(k0 + 32 + (tid >> 4)) * K3 + n0 + (tid & 15) * 4, 0, 3);
    }
    __syncthreads();

    const v16h a0 = frag_a_ld(&As[wm][0], 40);
    const v16h a1 = frag_a_ld(&As[wm + 16][0], 40);
    const v16h b0 = frag_b_ld(&Bt[wn][0], 40);
    const v16h b1 = frag_b_ld(&Bt[wn + 16][0], 40);
    acc[0] = wmma_f16(a0, b0, acc[0]);
    acc[1] = wmma_f16(a0, b1, acc[1]);
    acc[2] = wmma_f16(a1, b0, acc[2]);
    acc[3] = wmma_f16(a1, b1, acc[3]);
  }

  // C layout: lane holds col (lane&15); reg r -> row r + 8*(lane>=16).
  const int rhi = (lane >> 4) << 3;
  const int cl  = lane & 15;
#pragma unroll
  for (int t = 0; t < 4; ++t) {
    const int rb = m0 + wm + (t >> 1) * 16 + rhi;
    const int cb = n0 + wn + (t & 1) * 16 + cl;
    const bool relu = cb < 2 * CDIM;  // q and k regions
#pragma unroll
    for (int r = 0; r < 8; ++r) {
      float v = acc[t][r];
      if (relu) v = fmaxf(v, 0.0f);
      qkvh[(size_t)(rb + r) * K3 + cb] = (_Float16)v;
    }
  }
}

// ---------------------------------------------------------------------------
// Kernel 2: flash attention. Block = (head h, 128 query rows), 8 waves,
// each wave owns 16 rows. Loop over keys in 32-chunks with online softmax.
// K fragments come straight from global (row-major K == B^T layout).
// ---------------------------------------------------------------------------
__global__ __launch_bounds__(256) void attn_kernel(
    const _Float16* __restrict__ qkvh, _Float16* __restrict__ attnh) {
  __shared__ __align__(16) _Float16 Vt[32][40];      // V^T chunk: [d][j]
  __shared__ __align__(16) _Float16 Ps[8][16][40];   // per-wave P tile

  const int tid  = threadIdx.x;
  const int lane = tid & 31;
  const int wid  = tid >> 5;
  const int h    = blockIdx.y;
  const int m0   = blockIdx.x * 128;

  // Q fragment (A layout), 1/sqrt(DH) folded in.
  v16h qa;
  {
    const int row = m0 + wid * 16 + (lane & 15);
    const int khi = (lane >> 4) << 3;
    const _Float16* rp = qkvh + (size_t)row * K3 + h * DHEAD;
#pragma unroll
    for (int i = 0; i < 8; ++i) {
      const int k = ((i < 4) ? (2 * i) : (8 + 2 * i)) + khi;
      qa[2 * i]     = (_Float16)((float)rp[k]     * QKSCALE);
      qa[2 * i + 1] = (_Float16)((float)rp[k + 1] * QKSCALE);
    }
  }

  v8f O0 = {}, O1 = {};
  float mrun[8], lrun[8];
#pragma unroll
  for (int r = 0; r < 8; ++r) { mrun[r] = -__builtin_inff(); lrun[r] = 0.0f; }

  for (int j0 = 0; j0 < NTOK; j0 += 32) {
    __syncthreads();
    // stage V^T: Vt[d][j] = V[j0+j][d]
    {
      const int j = tid >> 3;
      const int d = (tid & 7) * 4;
      const _Float16* vp = qkvh + (size_t)(j0 + j) * K3 + 2 * CDIM + h * DHEAD + d;
#pragma unroll
      for (int q = 0; q < 4; ++q) Vt[d + q][j] = vp[q];
      // prefetch next chunk's K and V rows
      if (j0 + 32 < NTOK) {
        __builtin_prefetch(qkvh + (size_t)(j0 + 32 + j) * K3 + CDIM + h * DHEAD, 0, 3);
        __builtin_prefetch(qkvh + (size_t)(j0 + 32 + j) * K3 + 2 * CDIM + h * DHEAD, 0, 3);
      }
    }
    __syncthreads();

    // S = (Q*scale) @ K^T for keys j0..j0+31 (K fragments direct from global)
    const _Float16* kp = qkvh + (size_t)j0 * K3 + CDIM + h * DHEAD;
    const v16h bk0 = frag_b_ld(kp, K3);
    const v16h bk1 = frag_b_ld(kp + (size_t)16 * K3, K3);
    v8f S0 = {}, S1 = {};
    S0 = wmma_f16(qa, bk0, S0);
    S1 = wmma_f16(qa, bk1, S1);

    // Online softmax: reduce across the 16 lanes sharing each row
    // (xor masks 1,2,4,8 stay inside a 16-lane half).
#pragma unroll
    for (int r = 0; r < 8; ++r) {
      float cm = fmaxf(S0[r], S1[r]);
      cm = fmaxf(cm, __shfl_xor(cm, 1, 32));
      cm = fmaxf(cm, __shfl_xor(cm, 2, 32));
      cm = fmaxf(cm, __shfl_xor(cm, 4, 32));
      cm = fmaxf(cm, __shfl_xor(cm, 8, 32));
      const float mn    = fmaxf(mrun[r], cm);
      const float alpha = __expf(mrun[r] - mn);
      const float p0 = __expf(S0[r] - mn);
      const float p1 = __expf(S1[r] - mn);
      float rs = p0 + p1;
      rs += __shfl_xor(rs, 1, 32);
      rs += __shfl_xor(rs, 2, 32);
      rs += __shfl_xor(rs, 4, 32);
      rs += __shfl_xor(rs, 8, 32);
      lrun[r] = lrun[r] * alpha + rs;
      mrun[r] = mn;
      O0[r] *= alpha;
      O1[r] *= alpha;
      // C layout -> A layout reshuffle through wave-private LDS
      const int rl = r + ((lane >> 4) << 3);
      const int cl = lane & 15;
      Ps[wid][rl][cl]      = (_Float16)p0;
      Ps[wid][rl][cl + 16] = (_Float16)p1;
    }

    const v16h pa  = frag_a_ld(&Ps[wid][0][0], 40);
    const v16h bv0 = frag_b_ld(&Vt[0][0], 40);   // dims 0..15
    const v16h bv1 = frag_b_ld(&Vt[16][0], 40);  // dims 16..31
    O0 = wmma_f16(pa, bv0, O0);
    O1 = wmma_f16(pa, bv1, O1);
  }

  const int rhi = (lane >> 4) << 3;
  const int cl  = lane & 15;
#pragma unroll
  for (int r = 0; r < 8; ++r) {
    const int row = m0 + wid * 16 + r + rhi;
    const float inv = 1.0f / lrun[r];
    attnh[(size_t)row * CDIM + h * DHEAD + cl]      = (_Float16)(O0[r] * inv);
    attnh[(size_t)row * CDIM + h * DHEAD + cl + 16] = (_Float16)(O1[r] * inv);
  }
}

// ---------------------------------------------------------------------------
// Kernel 3: out[4096,512] = attn_f16 @ Wout[512,512] + b_out (f32 out)
// ---------------------------------------------------------------------------
__global__ __launch_bounds__(256) void out_gemm_kernel(
    const _Float16* __restrict__ attnh, const float* __restrict__ Wout,
    const float* __restrict__ bias, float* __restrict__ out) {
  __shared__ __align__(16) _Float16 As[128][40];
  __shared__ __align__(16) _Float16 Bt[64][40];

  const int tid  = threadIdx.x;
  const int lane = tid & 31;
  const int wid  = tid >> 5;
  const int m0   = blockIdx.y * 128;
  const int n0   = blockIdx.x * 64;
  const int wm   = (wid >> 1) * 32;
  const int wn   = (wid & 1) * 32;

  v8f acc[4] = {};

  for (int k0 = 0; k0 < CDIM; k0 += 32) {
    __syncthreads();
#pragma unroll
    for (int i = 0; i < 4; ++i) {
      const int idx = tid + i * 256;
      const int r = idx >> 3, c = (idx & 7) * 4;
      *(uint2*)&As[r][c] = *(const uint2*)(attnh + (size_t)(m0 + r) * CDIM + k0 + c);
    }
#pragma unroll
    for (int i = 0; i < 2; ++i) {
      const int idx = tid + i * 256;
      const int kr = idx >> 4, c = (idx & 15) * 4;
      const float4 v = *(const float4*)(Wout + (size_t)(k0 + kr) * CDIM + n0 + c);
      Bt[c][kr]     = (_Float16)v.x;
      Bt[c + 1][kr] = (_Float16)v.y;
      Bt[c + 2][kr] = (_Float16)v.z;
      Bt[c + 3][kr] = (_Float16)v.w;
    }
    if (k0 + 32 < CDIM) {
      __builtin_prefetch(attnh + (size_t)(m0 + (tid >> 3)) * CDIM + (k0 + 32) + (tid & 7) * 4, 0, 3);
      __builtin_prefetch(Wout + (size_t)(k0 + 32 + (tid >> 4)) * CDIM + n0 + (tid & 15) * 4, 0, 3);
    }
    __syncthreads();

    const v16h a0 = frag_a_ld(&As[wm][0], 40);
    const v16h a1 = frag_a_ld(&As[wm + 16][0], 40);
    const v16h b0 = frag_b_ld(&Bt[wn][0], 40);
    const v16h b1 = frag_b_ld(&Bt[wn + 16][0], 40);
    acc[0] = wmma_f16(a0, b0, acc[0]);
    acc[1] = wmma_f16(a0, b1, acc[1]);
    acc[2] = wmma_f16(a1, b0, acc[2]);
    acc[3] = wmma_f16(a1, b1, acc[3]);
  }

  const int rhi = (lane >> 4) << 3;
  const int cl  = lane & 15;
#pragma unroll
  for (int t = 0; t < 4; ++t) {
    const int rb = m0 + wm + (t >> 1) * 16 + rhi;
    const int cb = n0 + wn + (t & 1) * 16 + cl;
    const float bv = bias[cb];
#pragma unroll
    for (int r = 0; r < 8; ++r)
      out[(size_t)(rb + r) * CDIM + cb] = acc[t][r] + bv;
  }
}

// ---------------------------------------------------------------------------
extern "C" void kernel_launch(void* const* d_in, const int* in_sizes, int n_in,
                              void* d_out, int out_size, void* d_ws, size_t ws_size,
                              hipStream_t stream) {
  const float* x    = (const float*)d_in[0];
  const float* Wqkv = (const float*)d_in[1];
  const float* Wout = (const float*)d_in[2];
  const float* bias = (const float*)d_in[3];
  float* out = (float*)d_out;

  char* ws = (char*)d_ws;
  _Float16* qkvh  = (_Float16*)ws;                              // [4096][1536] f16
  _Float16* attnh = (_Float16*)(ws + (size_t)NTOK * K3 * 2);    // [4096][512]  f16

  qkv_gemm_kernel<<<dim3(K3 / 64, NTOK / 128), 256, 0, stream>>>(x, Wqkv, qkvh);
  attn_kernel<<<dim3(NTOK / 128, HEADS), 256, 0, stream>>>(qkvh, attnh);
  out_gemm_kernel<<<dim3(CDIM / 64, NTOK / 128), 256, 0, stream>>>(attnh, Wout, bias, out);

  (void)in_sizes; (void)n_in; (void)out_size; (void)ws_size;
}
